// HeteroGAT3_29807073034444
// MI455X (gfx1250) — compile-verified, matrix operand order checked by reading, not run on hardware
//
#include <hip/hip_runtime.h>
#include <hip/hip_bf16.h>
#include <math.h>

#define NPN 100000
#define NDN 10000
#define NE  500000
#define KIN 128
#define HH  4
#define CCH 32
#define FF  128
#define CH2 64
#define OUTD 10

typedef __attribute__((ext_vector_type(16))) __bf16 bf16x16;
typedef __attribute__((ext_vector_type(8)))  float  f32x8;

// ---------- atomics ----------
__device__ __forceinline__ void atomic_add_f32(float* p, float v) {
  unsafeAtomicAdd(p, v);   // -> global_atomic_add_f32
}
__device__ __forceinline__ void atomic_max_f32(float* addr, float value) {
  // order-preserving int trick (no NaNs in this workload)
  if (value >= 0.f) atomicMax((int*)addr, __float_as_int(value));
  else              atomicMin((unsigned int*)addr, __float_as_uint(value));
}

// ---------- utility kernels ----------
__global__ void fill_f32(float* __restrict__ p, float v, int n) {
  int i = blockIdx.x * blockDim.x + threadIdx.x;
  if (i < n) p[i] = v;
}
__global__ void init_bias(float* __restrict__ out, const float* __restrict__ b, int Nn, int F) {
  int i = blockIdx.x * blockDim.x + threadIdx.x;
  if (i < Nn * F) out[i] = b[i % F];
}

// ---------- WMMA GEMM: C = act(A[MxK] @ W[KxN] + bias + addend) ----------
// 256 threads = 8 waves; each wave owns one 16x16 tile; all waves share ntile
// so the W tile is staged once per block into LDS in B-lane layout (bf16),
// then each lane reads 32B contiguous (ds_load_b128 x2) per k-step.
template<int K, bool BIAS, bool ADD, bool ACT>
__global__ __launch_bounds__(256) void gemm_bf16_wmma(
    const float* __restrict__ A, const float* __restrict__ W,
    const float* __restrict__ bias, const float* __restrict__ addend,
    float* __restrict__ Cmat, int M, int N) {
  constexpr int KS = K / 32;
  __shared__ __align__(32) __bf16 ldsB[KS][32][16];
  const int colbase = blockIdx.y * 16;
  // cooperative stage of W tile (K x 16) into B-lane-swizzled LDS, fp32->bf16
  for (int idx = threadIdx.x; idx < KS * 32 * 16; idx += 256) {
    const int e = idx & 15;
    const int l = (idx >> 4) & 31;
    const int s = idx >> 9;
    const int k = s * 32 + ((l >> 4) << 3) + (e < 8 ? e : e + 8);
    ldsB[s][l][e] = (__bf16)W[(size_t)k * N + colbase + (l & 15)];
  }
  __syncthreads();

  const int wave  = threadIdx.x >> 5;
  const int lane  = threadIdx.x & 31;
  const int mtile = blockIdx.x * 8 + wave;
  if (mtile * 16 >= M) return;                // uniform per wave
  const int hi  = lane >> 4;
  const int l16 = lane & 15;
  int row = mtile * 16 + l16; if (row >= M) row = M - 1;
  const float* arow = A + (size_t)row * K;

  f32x8 acc = {0.f, 0.f, 0.f, 0.f, 0.f, 0.f, 0.f, 0.f};
#pragma unroll
  for (int s = 0; s < KS; ++s) {
    const int kb = s * 32 + (hi << 3);        // ISA 16-bit A lane K mapping
    float av[16];
    *(float4*)(av + 0)  = *(const float4*)(arow + kb);
    *(float4*)(av + 4)  = *(const float4*)(arow + kb + 4);
    *(float4*)(av + 8)  = *(const float4*)(arow + kb + 16);
    *(float4*)(av + 12) = *(const float4*)(arow + kb + 20);
    bf16x16 a;
#pragma unroll
    for (int e = 0; e < 16; ++e) a[e] = (__bf16)av[e];
    bf16x16 b = *(const bf16x16*)(&ldsB[s][lane][0]);
    acc = __builtin_amdgcn_wmma_f32_16x16x32_bf16(false, a, false, b,
                                                  (short)0, acc, false, false);
  }

  const int col   = colbase + l16;
  const int rbase = mtile * 16 + (hi << 3);   // C/D layout: vgpr j -> row rbase+j
  float bv = 0.f;
  if (BIAS) bv = bias[col];
#pragma unroll
  for (int j = 0; j < 8; ++j) {
    const int r = rbase + j;
    float v = acc[j] + bv;
    if (ADD) v += addend[(size_t)r * N + col];
    if (ACT) v = v > 0.f ? v : expm1f(v);     // ELU(alpha=1)
    Cmat[(size_t)r * N + col] = v;
  }
}

// ---------- al_s[n,h] = sum_c hs[n,h,c]*att[h,c] ----------
__global__ void attn_reduce(const float* __restrict__ hs, const float* __restrict__ att,
                            float* __restrict__ al, int Nn, int H, int C) {
  int i = blockIdx.x * blockDim.x + threadIdx.x;
  if (i >= Nn * H) return;
  int n = i / H, h = i % H;
  const float* p = hs + (size_t)n * (H * C) + h * C;
  const float* a = att + h * C;
  float acc = 0.f;
  for (int c = 0; c < C; ++c) acc += p[c] * a[c];
  al[i] = acc;
}

// ---------- aw[f,h] = sum_c Wdst[f,h,c]*attd[h,c]  (folds hd projection) ----------
__global__ void make_aw(const float* __restrict__ Wd, const float* __restrict__ attd,
                        float* __restrict__ aw, int K, int H, int C) {
  int i = blockIdx.x * blockDim.x + threadIdx.x;
  if (i >= K * H) return;
  int f = i / H, h = i % H;
  const float* wrow = Wd + (size_t)f * (H * C) + h * C;
  const float* a = attd + h * C;
  float acc = 0.f;
  for (int c = 0; c < C; ++c) acc += wrow[c] * a[c];
  aw[f * H + h] = acc;
}

// ---------- al_d[n,h] = sum_f X[n,f]*aw[f,h] ----------
__global__ void matvec_al(const float* __restrict__ X, const float* __restrict__ aw,
                          float* __restrict__ al, int Nn, int K, int H) {
  int i = blockIdx.x * blockDim.x + threadIdx.x;
  if (i >= Nn * H) return;
  int n = i / H, h = i % H;
  const float* x = X + (size_t)n * K;
  float acc = 0.f;
  for (int f = 0; f < K; ++f) acc += x[f] * aw[f * H + h];
  al[i] = acc;
}

// ---------- edge softmax passes (templated on H: shifts instead of div) ----------
template<int H>
__global__ void edge_pass1(const int* __restrict__ src, const int* __restrict__ dst,
                           const float* __restrict__ alS, const float* __restrict__ alD,
                           float* __restrict__ m, float* __restrict__ ebuf, int E) {
  int i = blockIdx.x * blockDim.x + threadIdx.x;
  if (i >= E * H) return;
  int e = (H == 1) ? i : (i / H);
  int h = (H == 1) ? 0 : (i % H);
  float v = alS[src[e] * H + h] + alD[dst[e] * H + h];
  v = v > 0.f ? v : 0.2f * v;                 // leaky_relu(0.2)
  ebuf[i] = v;
  atomic_max_f32(&m[dst[e] * H + h], v);
}
template<int H>
__global__ void edge_pass2(const int* __restrict__ dst, const float* __restrict__ m,
                           float* __restrict__ s, float* __restrict__ ebuf, int E) {
  int i = blockIdx.x * blockDim.x + threadIdx.x;
  if (i >= E * H) return;
  int e = (H == 1) ? i : (i / H);
  int h = (H == 1) ? 0 : (i % H);
  float p = expf(ebuf[i] - m[dst[e] * H + h]);
  ebuf[i] = p;
  atomic_add_f32(&s[dst[e] * H + h], p);
}
// one wave per edge; each lane handles F/32 channels (vector gather + f32 atomics)
template<int H, int C>
__global__ void edge_pass3(const int* __restrict__ src, const int* __restrict__ dst,
                           const float* __restrict__ hs, const float* __restrict__ ebuf,
                           const float* __restrict__ s, float* __restrict__ out, int E) {
  constexpr int F = H * C;
  constexpr int CPL = F >> 5;
  int edge = blockIdx.x * (blockDim.x >> 5) + (threadIdx.x >> 5);
  int lane = threadIdx.x & 31;
  if (edge >= E) return;
  int sn = src[edge], dn = dst[edge];
  int c0 = lane * CPL;
  int h = c0 / C;
  float alpha = ebuf[(size_t)edge * H + h] / (s[dn * H + h] + 1e-16f);
  const float* hrow = hs + (size_t)sn * F + c0;
  float* orow = out + (size_t)dn * F + c0;
  float hv[CPL];
  if (CPL == 4)      *(float4*)hv = *(const float4*)hrow;
  else if (CPL == 2) *(float2*)hv = *(const float2*)hrow;
#pragma unroll
  for (int c = 0; c < CPL; ++c) atomic_add_f32(&orow[c], hv[c] * alpha);
}

// ---------- lin3 + log_softmax ----------
__global__ void lin3_lsm(const float* __restrict__ X, const float* __restrict__ W,
                         const float* __restrict__ b, float* __restrict__ out, int M) {
  int r = blockIdx.x * blockDim.x + threadIdx.x;
  if (r >= M) return;
  const float* x = X + (size_t)r * CH2;
  float lg[OUTD];
#pragma unroll
  for (int j = 0; j < OUTD; ++j) lg[j] = b[j];
  for (int k = 0; k < CH2; ++k) {
    float xv = x[k];
#pragma unroll
    for (int j = 0; j < OUTD; ++j) lg[j] += xv * W[k * OUTD + j];
  }
  float mx = lg[0];
#pragma unroll
  for (int j = 1; j < OUTD; ++j) mx = fmaxf(mx, lg[j]);
  float sum = 0.f;
#pragma unroll
  for (int j = 0; j < OUTD; ++j) sum += expf(lg[j] - mx);
  float lse = mx + logf(sum);
#pragma unroll
  for (int j = 0; j < OUTD; ++j) out[(size_t)r * OUTD + j] = lg[j] - lse;
}

// ---------- host orchestration ----------
extern "C" void kernel_launch(void* const* d_in, const int* in_sizes, int n_in,
                              void* d_out, int out_size, void* d_ws, size_t ws_size,
                              hipStream_t stream) {
  const float* x_person     = (const float*)d_in[0];
  const float* x_diag       = (const float*)d_in[1];
  const float* conv_Wsrc    = (const float*)d_in[3];
  const float* conv_Wdst    = (const float*)d_in[4];
  const float* conv_att_src = (const float*)d_in[5];
  const float* conv_att_dst = (const float*)d_in[6];
  const float* conv_bias    = (const float*)d_in[7];
  const float* g2_Wsrc      = (const float*)d_in[8];
  const float* g2_Wdst      = (const float*)d_in[9];
  const float* g2_att_src   = (const float*)d_in[10];
  const float* g2_att_dst   = (const float*)d_in[11];
  const float* g2_bias      = (const float*)d_in[12];
  const float* lin1_W       = (const float*)d_in[13];
  const float* lin1_b       = (const float*)d_in[14];
  const float* lin2_W       = (const float*)d_in[15];
  const float* lin2_b       = (const float*)d_in[16];
  const float* lin3_W       = (const float*)d_in[17];
  const float* lin3_b       = (const float*)d_in[18];
  const int* p2d_src = (const int*)d_in[23];
  const int* p2d_dst = (const int*)d_in[24];
  const int* d2p_src = (const int*)d_in[25];
  const int* d2p_dst = (const int*)d_in[26];

  float* w = (float*)d_ws;
  size_t o = 0;
  auto alloc = [&](size_t n) { float* p = w + o; o += n; return p; };
  float* hsP  = alloc((size_t)NPN * FF);   // person-sized hs
  float* outP = alloc((size_t)NPN * FF);   // person-sized accumulator (npn)
  float* xpN  = alloc((size_t)NPN * FF);   // next xp
  float* hsD  = alloc((size_t)NDN * FF);
  float* outD = alloc((size_t)NDN * FF);
  float* xdN  = alloc((size_t)NDN * FF);
  float* alS  = alloc((size_t)NPN * HH);
  float* alD  = alloc((size_t)NPN * HH);
  float* mbuf = alloc((size_t)NPN * HH);
  float* sbuf = alloc((size_t)NPN * HH);
  float* ebuf = alloc((size_t)NE  * HH);
  float* awb  = alloc((size_t)KIN * HH);
  (void)ws_size; (void)in_sizes; (void)n_in; (void)out_size;

  auto cdiv = [](int a, int b) { return (a + b - 1) / b; };
  // gemm variants: hs projection (no epilogue), residual+bias+ELU (K=128 / K=64)
  auto gemm_hs = [&](const float* A, const float* Wm, float* Cm, int M, int N) {
    dim3 grd(cdiv(M / 16, 8), N / 16);
    gemm_bf16_wmma<KIN, false, false, false><<<grd, 256, 0, stream>>>(
        A, Wm, nullptr, nullptr, Cm, M, N);
  };
  auto gemm_res128 = [&](const float* A, const float* Wm, const float* bias,
                         const float* add, float* Cm, int M) {
    dim3 grd(cdiv(M / 16, 8), FF / 16);
    gemm_bf16_wmma<FF, true, true, true><<<grd, 256, 0, stream>>>(
        A, Wm, bias, add, Cm, M, FF);
  };
  auto gemm_res64 = [&](const float* A, const float* Wm, const float* bias,
                        const float* add, float* Cm, int M) {
    dim3 grd(cdiv(M / 16, 8), CH2 / 16);
    gemm_bf16_wmma<CH2, true, true, true><<<grd, 256, 0, stream>>>(
        A, Wm, bias, add, Cm, M, CH2);
  };

  // GAT for the conv layers (H=4, C=32)
  auto run_gat4 = [&](const float* xs, const float* xdst, int Ns, int Nd,
                      const int* src, const int* dst,
                      const float* Wsrc, const float* Wdst,
                      const float* attS, const float* attD, const float* bias,
                      float* hsbuf, float* outbuf) {
    gemm_hs(xs, Wsrc, hsbuf, Ns, FF);
    attn_reduce<<<cdiv(Ns * HH, 256), 256, 0, stream>>>(hsbuf, attS, alS, Ns, HH, CCH);
    make_aw<<<1, 512, 0, stream>>>(Wdst, attD, awb, KIN, HH, CCH);
    matvec_al<<<cdiv(Nd * HH, 256), 256, 0, stream>>>(xdst, awb, alD, Nd, KIN, HH);
    fill_f32<<<cdiv(Nd * HH, 256), 256, 0, stream>>>(mbuf, -__builtin_huge_valf(), Nd * HH);
    fill_f32<<<cdiv(Nd * HH, 256), 256, 0, stream>>>(sbuf, 0.f, Nd * HH);
    init_bias<<<cdiv(Nd * FF, 256), 256, 0, stream>>>(outbuf, bias, Nd, FF);
    edge_pass1<HH><<<cdiv(NE * HH, 256), 256, 0, stream>>>(src, dst, alS, alD, mbuf, ebuf, NE);
    edge_pass2<HH><<<cdiv(NE * HH, 256), 256, 0, stream>>>(dst, mbuf, sbuf, ebuf, NE);
    edge_pass3<HH, CCH><<<cdiv(NE, 8), 256, 0, stream>>>(src, dst, hsbuf, ebuf, sbuf, outbuf, NE);
  };

  const float* xp = x_person;
  const float* xd = x_diag;
  for (int l = 0; l < 2; ++l) {
    // p2d: src=person, dst=diag
    run_gat4(xp, xd, NPN, NDN, p2d_src, p2d_dst,
             conv_Wsrc + (size_t)(l * 2 + 0) * KIN * FF, conv_Wdst + (size_t)(l * 2 + 0) * KIN * FF,
             conv_att_src + (l * 2 + 0) * FF, conv_att_dst + (l * 2 + 0) * FF,
             conv_bias + (l * 2 + 0) * FF, hsP, outD);
    // d2p: src=diag, dst=person
    run_gat4(xd, xp, NDN, NPN, d2p_src, d2p_dst,
             conv_Wsrc + (size_t)(l * 2 + 1) * KIN * FF, conv_Wdst + (size_t)(l * 2 + 1) * KIN * FF,
             conv_att_src + (l * 2 + 1) * FF, conv_att_dst + (l * 2 + 1) * FF,
             conv_bias + (l * 2 + 1) * FF, hsD, outP);
    // x = elu(n + n@lin1_W + lin1_b)
    gemm_res128(outP, lin1_W, lin1_b, outP, xpN, NPN);
    gemm_res128(outD, lin1_W, lin1_b, outD, xdN, NDN);
    xp = xpN; xd = xdN;
  }

  // final d2p GAT, heads=1, ch=64 (final p2d GAT + edge MLP are dead -> skipped)
  {
    gemm_hs(xd, g2_Wsrc + (size_t)1 * KIN * CH2, hsD, NDN, CH2);
    attn_reduce<<<cdiv(NDN, 256), 256, 0, stream>>>(hsD, g2_att_src + CH2, alS, NDN, 1, CH2);
    make_aw<<<1, 128, 0, stream>>>(g2_Wdst + (size_t)1 * KIN * CH2, g2_att_dst + CH2, awb, KIN, 1, CH2);
    matvec_al<<<cdiv(NPN, 256), 256, 0, stream>>>(xp, awb, alD, NPN, KIN, 1);
    fill_f32<<<cdiv(NPN, 256), 256, 0, stream>>>(mbuf, -__builtin_huge_valf(), NPN);
    fill_f32<<<cdiv(NPN, 256), 256, 0, stream>>>(sbuf, 0.f, NPN);
    init_bias<<<cdiv(NPN * CH2, 256), 256, 0, stream>>>(outP, g2_bias + CH2, NPN, CH2);
    edge_pass1<1><<<cdiv(NE, 256), 256, 0, stream>>>(d2p_src, d2p_dst, alS, alD, mbuf, ebuf, NE);
    edge_pass2<1><<<cdiv(NE, 256), 256, 0, stream>>>(d2p_dst, mbuf, sbuf, ebuf, NE);
    edge_pass3<1, CH2><<<cdiv(NE, 8), 256, 0, stream>>>(d2p_src, d2p_dst, hsD, ebuf, sbuf, outP, NE);
  }
  // xp = elu(npn + npn@lin2_W + lin2_b)   (reuse xpN as NPN x 64)
  gemm_res64(outP, lin2_W, lin2_b, outP, xpN, NPN);
  // out = log_softmax(xp@lin3_W + lin3_b)
  lin3_lsm<<<cdiv(NPN, 256), 256, 0, stream>>>(xpN, lin3_W, lin3_b, (float*)d_out, NPN);
}
